// GCNRefinementHead_37194416783772
// MI455X (gfx1250) — matrix operand order repaired and checked by reading, not compiled, not run on hardware
//
#include <hip/hip_runtime.h>
#include <math.h>

// ---- problem constants (match reference) ----
#define NJ      21
#define NBATCH  1024
#define CCH     1280
#define FH      16
#define FW      12
#define HID     512
#define NLAY    4
#define KIN     1283          // C + 3
#define NTK0    41            // ceil(1312/32): padded K for input layer
#define NTK     16            // 512/32 k-tiles for GCN layers
#define NTN     32            // 512/16 n-tiles
#define LN_EPS  1e-5f

typedef __attribute__((ext_vector_type(16))) __bf16 v16bf;
typedef __attribute__((ext_vector_type(8)))  float  v8f;

// fragment element -> K index mappings per CDNA5 16-bit WMMA VGPR layouts
// A (16x32): lanes 0-15 hold M=lane, K = {0..7,16..23}; lanes 16-31 hold K = {8..15,24..31}
__device__ __forceinline__ int a_k_of(int lane, int j) {
    return (j & 7) + ((j >> 3) << 4) + ((lane >> 4) << 3);
}
// B (32x16): lanes 0-15 col N=lane K = {0..7,16..23}; lanes 16-31 col N=lane-16 K = {8..15,24..31}
__device__ __forceinline__ int b_k_of(int lane, int j) {
    return (j & 7) + ((j >> 3) << 4) + ((lane >> 4) << 3);
}

__device__ __forceinline__ v8f v8f_zero() {
    v8f z = {0.f, 0.f, 0.f, 0.f, 0.f, 0.f, 0.f, 0.f};
    return z;
}

// ---------------------------------------------------------------------------
// Prep: convert w_in (1283x512 f32, K zero-padded to 1312) into bf16 fragments
// layout: [kt(41)][nt(32)][lane(32)][j(16)]
// ---------------------------------------------------------------------------
__global__ void prep_win(const float* __restrict__ w_in, __bf16* __restrict__ wf) {
    const int total = NTK0 * NTN * 32 * 16;
    for (int idx = blockIdx.x * blockDim.x + threadIdx.x; idx < total;
         idx += gridDim.x * blockDim.x) {
        int j    = idx & 15;
        int lane = (idx >> 4) & 31;
        int nt   = (idx >> 9) & 31;
        int kt   = idx >> 14;
        int k = kt * 32 + b_k_of(lane, j);
        int n = nt * 16 + (lane & 15);
        float v = (k < KIN) ? w_in[k * HID + n] : 0.f;
        wf[idx] = (__bf16)v;
    }
}

// w_gcn (4 x 512 x 512) -> bf16 fragments: [l][kt(16)][nt(32)][lane][j]
__global__ void prep_wgcn(const float* __restrict__ w_gcn, __bf16* __restrict__ wf) {
    const int total = NLAY * NTK * NTN * 32 * 16;
    for (int idx = blockIdx.x * blockDim.x + threadIdx.x; idx < total;
         idx += gridDim.x * blockDim.x) {
        int j    = idx & 15;
        int lane = (idx >> 4) & 31;
        int nt   = (idx >> 9) & 31;
        int kt   = (idx >> 14) & 15;
        int l    = idx >> 18;
        int k = kt * 32 + b_k_of(lane, j);
        int n = nt * 16 + (lane & 15);
        wf[idx] = (__bf16)w_gcn[(l * HID + k) * HID + n];
    }
}

// ---------------------------------------------------------------------------
// Fused per-batch kernel: grid-sample gather -> in-proj GEMM -> LN+GELU ->
// 4x (A_hat mix -> GEMM -> LN+GELU) -> 512->3 head + residual.
// One block (128 threads = 4 waves) per batch element.
// ---------------------------------------------------------------------------

// dynamic LDS layout (bytes)
#define OFF_A   0                      // aBuf  : 16kt x 2mt x 32lane x 16 bf16 = 32768 B
#define OFF_H   32768                  // Hbuf  : 32 x 512 f32                 = 65536 B
#define OFF_G   (32768 + 65536)        // Hg    : 32 x 512 bf16                = 32768 B
#define OFF_AH  (OFF_G + 32768)        // A_hat : 441 f32 (pad 1792)
#define OFF_PI  (OFF_AH + 1792)        // joint corner ints: 21*4 (pad 352)
#define OFF_PW  (OFF_PI + 352)         // joint tap weights: 21*4 f32 (pad 352)
#define SMEM_BYTES (OFF_PW + 352)

__global__ void __launch_bounds__(128, 1)
gcn_head_fused(const float* __restrict__ joints,
               const float* __restrict__ feat,
               const float* __restrict__ kp2d,
               const float* __restrict__ A_hat,
               const float* __restrict__ b_in,
               const float* __restrict__ g_in,
               const float* __restrict__ be_in,
               const float* __restrict__ b_gcn,
               const float* __restrict__ g_gcn,
               const float* __restrict__ be_gcn,
               const float* __restrict__ w_d,
               const float* __restrict__ b_d,
               const __bf16* __restrict__ wf_in,
               const __bf16* __restrict__ wf_gcn,
               float* __restrict__ out) {
    extern __shared__ char smem[];
    __bf16* aBuf = (__bf16*)(smem + OFF_A);
    float*  Hbuf = (float*) (smem + OFF_H);
    __bf16* Hg   = (__bf16*)(smem + OFF_G);
    float*  Ahs  = (float*) (smem + OFF_AH);
    int*    pxi  = (int*)   (smem + OFF_PI);
    float*  pwf  = (float*) (smem + OFF_PW);

    const int b    = blockIdx.x;
    const int tid  = threadIdx.x;
    const int lane = tid & 31;
    const int wave = tid >> 5;

    // ---- stage: A_hat copy + per-joint bilinear params ----
    for (int i = tid; i < NJ * NJ; i += 128) Ahs[i] = A_hat[i];
    if (tid < NJ) {
        float u = kp2d[(b * NJ + tid) * 2 + 0] * (1.f / 0.375f);
        float v = kp2d[(b * NJ + tid) * 2 + 1] * (1.f / 0.5f);
        float ix = ((u + 1.f) * (float)FW - 1.f) * 0.5f;
        float iy = ((v + 1.f) * (float)FH - 1.f) * 0.5f;
        float x0 = floorf(ix), y0 = floorf(iy);
        float wx1 = ix - x0, wy1 = iy - y0;
        float wx0 = 1.f - wx1, wy0 = 1.f - wy1;
        int xi0 = (int)x0, yi0 = (int)y0, xi1 = xi0 + 1, yi1 = yi0 + 1;
        float mx0 = (xi0 >= 0 && xi0 < FW) ? 1.f : 0.f;
        float mx1 = (xi1 >= 0 && xi1 < FW) ? 1.f : 0.f;
        float my0 = (yi0 >= 0 && yi0 < FH) ? 1.f : 0.f;
        float my1 = (yi1 >= 0 && yi1 < FH) ? 1.f : 0.f;
        pwf[tid * 4 + 0] = wx0 * wy0 * mx0 * my0;
        pwf[tid * 4 + 1] = wx1 * wy0 * mx1 * my0;
        pwf[tid * 4 + 2] = wx0 * wy1 * mx0 * my1;
        pwf[tid * 4 + 3] = wx1 * wy1 * mx1 * my1;
        pxi[tid * 4 + 0] = min(max(xi0, 0), FW - 1);
        pxi[tid * 4 + 1] = min(max(xi1, 0), FW - 1);
        pxi[tid * 4 + 2] = min(max(yi0, 0), FH - 1);
        pxi[tid * 4 + 3] = min(max(yi1, 0), FH - 1);
    }
    __syncthreads();

    v8f acc[2][8];
#pragma unroll
    for (int mt = 0; mt < 2; ++mt)
#pragma unroll
        for (int nt = 0; nt < 8; ++nt) acc[mt][nt] = v8f_zero();

    // =========================== layer 0 (K = 1312, 3 chunks of k-tiles) ===
    for (int chunk = 0; chunk < 3; ++chunk) {
        const int ktCount = (chunk < 2) ? 16 : (NTK0 - 32);
        const int total = ktCount * 2 * 32 * 16;
        // pack A fragments: gathered node features, bf16
        for (int e = tid; e < total; e += 128) {
            int j   = e & 15;
            int ln  = (e >> 4) & 31;
            int mt  = (e >> 9) & 1;
            int ktL = e >> 10;
            int m = mt * 16 + (ln & 15);
            int k = (chunk * 16 + ktL) * 32 + a_k_of(ln, j);
            float val = 0.f;
            if (m < NJ) {
                if (k < CCH) {
                    const float* fb = feat + ((size_t)b * CCH + k) * (FH * FW);
                    int x0c = pxi[m * 4 + 0], x1c = pxi[m * 4 + 1];
                    int y0c = pxi[m * 4 + 2], y1c = pxi[m * 4 + 3];
                    val = pwf[m * 4 + 0] * fb[y0c * FW + x0c]
                        + pwf[m * 4 + 1] * fb[y0c * FW + x1c]
                        + pwf[m * 4 + 2] * fb[y1c * FW + x0c]
                        + pwf[m * 4 + 3] * fb[y1c * FW + x1c];
                } else if (k < KIN) {
                    val = joints[(b * NJ + m) * 3 + (k - CCH)];
                }
            }
            aBuf[((ktL * 2 + mt) * 32 + ln) * 16 + j] = (__bf16)val;
        }
        __syncthreads();
        // WMMA over this chunk's k-tiles
        for (int ktL = 0; ktL < ktCount; ++ktL) {
            int ktg = chunk * 16 + ktL;
            v16bf a0 = *(const v16bf*)&aBuf[((ktL * 2 + 0) * 32 + lane) * 16];
            v16bf a1 = *(const v16bf*)&aBuf[((ktL * 2 + 1) * 32 + lane) * 16];
            const __bf16* wb = wf_in + (size_t)ktg * NTN * 512;
            __builtin_prefetch(wb + (size_t)NTN * 512 + lane * 64, 0, 0); // next k-tile -> L2
#pragma unroll
            for (int nt = 0; nt < 8; ++nt) {
                int ntile = wave * 8 + nt;
                v16bf bf = *(const v16bf*)&wb[ntile * 512 + lane * 16];
                acc[0][nt] = __builtin_amdgcn_wmma_f32_16x16x32_bf16(
                    false, a0, false, bf, (short)0, acc[0][nt], false, false);
                acc[1][nt] = __builtin_amdgcn_wmma_f32_16x16x32_bf16(
                    false, a1, false, bf, (short)0, acc[1][nt], false, false);
            }
        }
        __syncthreads();
    }

    // store accumulators -> Hbuf (f32), then LN + GELU -> Hg (bf16)
    const float* bias = b_in;
    const float* gam  = g_in;
    const float* bet  = be_in;
    for (int layer = 0;; ++layer) {
        // C layout: VGPR e -> M = e (lanes 0-15) / e+8 (lanes 16-31), N = lane&15
#pragma unroll
        for (int mt = 0; mt < 2; ++mt)
#pragma unroll
            for (int nt = 0; nt < 8; ++nt) {
                int col = wave * 128 + nt * 16 + (lane & 15);
                int rb  = mt * 16 + ((lane >> 4) << 3);
#pragma unroll
                for (int e2 = 0; e2 < 8; ++e2)
                    Hbuf[(rb + e2) * HID + col] = acc[mt][nt][e2];
                acc[mt][nt] = v8f_zero();
            }
        __syncthreads();

        if (tid < NJ) {
            float s = 0.f;
            for (int k = 0; k < HID; ++k) s += Hbuf[tid * HID + k] + bias[k];
            float mu = s * (1.f / (float)HID);
            float vs = 0.f;
            for (int k = 0; k < HID; ++k) {
                float d = (Hbuf[tid * HID + k] + bias[k]) - mu;
                vs += d * d;
            }
            float rs = rsqrtf(vs * (1.f / (float)HID) + LN_EPS);
            for (int k = 0; k < HID; ++k) {
                float x = ((Hbuf[tid * HID + k] + bias[k]) - mu) * rs * gam[k] + bet[k];
                float g = 0.5f * x * (1.f + erff(x * 0.70710678118654752f));
                Hg[tid * HID + k] = (__bf16)g;
            }
        }
        __syncthreads();

        if (layer == NLAY) break;

        // pack next A fragments: (A_hat @ Hg) in bf16
        for (int e = tid; e < NTK * 2 * 32 * 16; e += 128) {
            int j   = e & 15;
            int ln  = (e >> 4) & 31;
            int mt  = (e >> 9) & 1;
            int ktL = e >> 10;
            int m = mt * 16 + (ln & 15);
            int k = ktL * 32 + a_k_of(ln, j);
            float val = 0.f;
            if (m < NJ) {
                for (int jj = 0; jj < NJ; ++jj)
                    val += Ahs[m * NJ + jj] * (float)Hg[jj * HID + k];
            }
            aBuf[((ktL * 2 + mt) * 32 + ln) * 16 + j] = (__bf16)val;
        }
        __syncthreads();

        const __bf16* wl = wf_gcn + (size_t)layer * NTK * NTN * 512;
        for (int kt = 0; kt < NTK; ++kt) {
            v16bf a0 = *(const v16bf*)&aBuf[((kt * 2 + 0) * 32 + lane) * 16];
            v16bf a1 = *(const v16bf*)&aBuf[((kt * 2 + 1) * 32 + lane) * 16];
            const __bf16* wb = wl + (size_t)kt * NTN * 512;
            __builtin_prefetch(wb + (size_t)NTN * 512 + lane * 64, 0, 0);
#pragma unroll
            for (int nt = 0; nt < 8; ++nt) {
                int ntile = wave * 8 + nt;
                v16bf bf = *(const v16bf*)&wb[ntile * 512 + lane * 16];
                acc[0][nt] = __builtin_amdgcn_wmma_f32_16x16x32_bf16(
                    false, a0, false, bf, (short)0, acc[0][nt], false, false);
                acc[1][nt] = __builtin_amdgcn_wmma_f32_16x16x32_bf16(
                    false, a1, false, bf, (short)0, acc[1][nt], false, false);
            }
        }
        __syncthreads();

        bias = b_gcn  + layer * HID;
        gam  = g_gcn  + layer * HID;
        bet  = be_gcn + layer * HID;
    }

    // head: delta = h @ w_d + b_d ; out = joints + delta
    if (tid < NJ * 3) {
        int r = tid / 3, c = tid % 3;
        float s = b_d[c];
        for (int k = 0; k < HID; ++k)
            s += (float)Hg[r * HID + k] * w_d[k * 3 + c];
        out[(b * NJ + r) * 3 + c] = joints[(b * NJ + r) * 3 + c] + s;
    }
}

// ---------------------------------------------------------------------------
extern "C" void kernel_launch(void* const* d_in, const int* in_sizes, int n_in,
                              void* d_out, int out_size, void* d_ws, size_t ws_size,
                              hipStream_t stream) {
    const float* joints = (const float*)d_in[0];
    const float* feat   = (const float*)d_in[1];
    const float* kp2d   = (const float*)d_in[2];
    const float* A_hat  = (const float*)d_in[3];
    const float* w_in   = (const float*)d_in[4];
    const float* b_in   = (const float*)d_in[5];
    const float* g_in   = (const float*)d_in[6];
    const float* be_in  = (const float*)d_in[7];
    const float* w_gcn  = (const float*)d_in[8];
    const float* b_gcn  = (const float*)d_in[9];
    const float* g_gcn  = (const float*)d_in[10];
    const float* be_gcn = (const float*)d_in[11];
    const float* w_d    = (const float*)d_in[12];
    const float* b_d    = (const float*)d_in[13];
    float* out = (float*)d_out;

    // workspace: bf16 weight fragments
    __bf16* wf_in  = (__bf16*)d_ws;
    const size_t win_elems = (size_t)NTK0 * NTN * 512;          // 671744
    __bf16* wf_gcn = (__bf16*)((char*)d_ws + win_elems * sizeof(__bf16));

    hipLaunchKernelGGL(prep_win, dim3((win_elems + 255) / 256), dim3(256), 0, stream,
                       w_in, wf_in);
    const size_t wg_elems = (size_t)NLAY * NTK * NTN * 512;     // 1048576
    hipLaunchKernelGGL(prep_wgcn, dim3((wg_elems + 255) / 256), dim3(256), 0, stream,
                       w_gcn, wf_gcn);

    (void)hipFuncSetAttribute((const void*)gcn_head_fused,
                              hipFuncAttributeMaxDynamicSharedMemorySize, SMEM_BYTES);
    hipLaunchKernelGGL(gcn_head_fused, dim3(NBATCH), dim3(128), SMEM_BYTES, stream,
                       joints, feat, kp2d, A_hat, b_in, g_in, be_in,
                       b_gcn, g_gcn, be_gcn, w_d, b_d, wf_in, wf_gcn, out);
}
